// FairGraphSAGE_76029511074008
// MI455X (gfx1250) — matrix operand order repaired and checked by reading, not compiled, not run on hardware
//
#include <hip/hip_runtime.h>
#include <hip/hip_bf16.h>

#define N_NODES 100000
#define N_EDGES 1000000
#define NODE_IN 64
#define EDGE_IN 64
#define HID 128
#define OUT_DIM 10

typedef __attribute__((ext_vector_type(16))) __bf16 bf16x16;
typedef __attribute__((ext_vector_type(8)))  __bf16 bf16x8;
typedef __attribute__((ext_vector_type(4)))  __bf16 bf16x4;
typedef __attribute__((ext_vector_type(8)))  float  f32x8;
typedef __attribute__((ext_vector_type(4)))  float  f32x4;

// ---------------------------------------------------------------------------
// WMMA helpers (CDNA5 wave32, V_WMMA_F32_16X16X32_BF16)
// ---------------------------------------------------------------------------
__device__ inline f32x8 zero8() {
    f32x8 z = {0.f, 0.f, 0.f, 0.f, 0.f, 0.f, 0.f, 0.f};
    return z;
}

__device__ inline f32x8 wmma_bf16(bf16x16 a, bf16x16 b, f32x8 c) {
    // (neg_a, A, neg_b, B, c_mod, C, reuse_a, reuse_b)
    return __builtin_amdgcn_wmma_f32_16x16x32_bf16(false, a, false, b,
                                                   (short)0, c, false, false);
}

__device__ inline bf16x16 combine16(bf16x8 lo, bf16x8 hi) {
    bf16x16 r;
#pragma unroll
    for (int i = 0; i < 8; ++i) { r[i] = lo[i]; r[i + 8] = hi[i]; }
    return r;
}

__device__ inline bf16x4 cvt4(f32x4 v) {
    bf16x4 r;
#pragma unroll
    for (int i = 0; i < 4; ++i) r[i] = (__bf16)v[i];
    return r;
}

// A-matrix fragment (16x32 bf16, row-major tile with leading dim `ld`):
// lane l: row = l&15, K = (l>>4)*8 + {0..7} and (l>>4)*8 + 16 + {0..7}
__device__ inline bf16x16 load_frag_a(const __bf16* base, int ld) {
    int lane = threadIdx.x & 31;
    const __bf16* p = base + (lane & 15) * ld + ((lane >> 4) << 3);
    bf16x8 lo = *(const bf16x8*)p;
    bf16x8 hi = *(const bf16x8*)(p + 16);
    return combine16(lo, hi);
}

// B-matrix fragment from a TRANSPOSED weight tile (stored [N][K], ld = K):
// lane l: col n = l&15, K = (l>>4)*8 + {0..7} and (l>>4)*8 + 16 + {0..7}
__device__ inline bf16x16 load_frag_b(const __bf16* base, int ld) {
    int lane = threadIdx.x & 31;
    const __bf16* p = base + (lane & 15) * ld + ((lane >> 4) << 3);
    bf16x8 lo = *(const bf16x8*)p;
    bf16x8 hi = *(const bf16x8*)(p + 16);
    return combine16(lo, hi);
}

__device__ inline void atom_add_f32(float* p, float v) {
    __hip_atomic_fetch_add(p, v, __ATOMIC_RELAXED, __HIP_MEMORY_SCOPE_AGENT);
}

// ---------------------------------------------------------------------------
// Weight prep: f32 [K][N] -> bf16 [Npad][K] (transposed, zero padded rows)
// ---------------------------------------------------------------------------
__global__ void transpose_bf16_kernel(const float* __restrict__ w,
                                      __bf16* __restrict__ wt,
                                      int K, int N, int Npad) {
    int idx = blockIdx.x * 256 + threadIdx.x;
    int total = Npad * K;
    if (idx >= total) return;
    int n = idx / K, k = idx - n * K;
    float v = (n < N) ? w[(size_t)k * N + n] : 0.f;
    wt[(size_t)n * K + k] = (__bf16)v;
}

__global__ void zero_kernel(float* __restrict__ p, int n) {
    int i = blockIdx.x * 256 + threadIdx.x;
    if (i < n) p[i] = 0.f;
}

// ---------------------------------------------------------------------------
// Degree / 1/max(deg,1)
// ---------------------------------------------------------------------------
__global__ void deg_kernel(const long long* __restrict__ dst, float* __restrict__ deg) {
    int e = blockIdx.x * 256 + threadIdx.x;
    if (e < N_EDGES) atom_add_f32(&deg[(int)dst[e]], 1.f);
}

__global__ void invd_kernel(const float* __restrict__ deg, float* __restrict__ invd) {
    int i = blockIdx.x * 256 + threadIdx.x;
    if (i < N_NODES) invd[i] = 1.f / fmaxf(deg[i], 1.f);
}

// ---------------------------------------------------------------------------
// Node encoder: h = relu(x @ w_node + b), bf16 out.
// M=32 per block (2 M-tiles per wave share each B fragment), 8 waves cover
// the 128 output columns. grid = N_NODES/32.
// ---------------------------------------------------------------------------
__global__ __launch_bounds__(256) void encoder_kernel(
    const float* __restrict__ x, const __bf16* __restrict__ wt,
    const float* __restrict__ bias, __bf16* __restrict__ hout) {
    __shared__ __align__(16) __bf16 sX[32 * NODE_IN];
    int t = threadIdx.x;
    int row0 = blockIdx.x * 32;
    const f32x4* xv = (const f32x4*)(x + (size_t)row0 * NODE_IN);
#pragma unroll
    for (int i = t; i < 32 * NODE_IN / 4; i += 256)
        ((bf16x4*)sX)[i] = cvt4(xv[i]);
    __syncthreads();
    int wave = t >> 5, lane = t & 31;
    int n0 = wave * 16;
    f32x8 acc[2] = {zero8(), zero8()};
#pragma unroll
    for (int kk = 0; kk < NODE_IN; kk += 32) {
        bf16x16 b = load_frag_b(wt + n0 * NODE_IN + kk, NODE_IN);
#pragma unroll
        for (int m = 0; m < 2; ++m)
            acc[m] = wmma_bf16(load_frag_a(sX + m * 16 * NODE_IN + kk, NODE_IN), b, acc[m]);
    }
    int col = lane & 15, rb = (lane >> 4) * 8;
    float bv = bias[n0 + col];
#pragma unroll
    for (int m = 0; m < 2; ++m)
#pragma unroll
        for (int r = 0; r < 8; ++r) {
            float v = fmaxf(acc[m][r] + bv, 0.f);
            hout[(size_t)(row0 + m * 16 + rb + r) * HID + n0 + col] = (__bf16)v;
        }
}

// ---------------------------------------------------------------------------
// Scatter: agg[dst] += h[src]  (one edge per wave; one b64 load + 4 f32
// atomics per lane). agg (51MB) is L2-resident at 192MB L2.
// ---------------------------------------------------------------------------
__global__ __launch_bounds__(256) void scatter_kernel(
    const long long* __restrict__ src, const long long* __restrict__ dst,
    const __bf16* __restrict__ h, float* __restrict__ agg) {
    int wave = threadIdx.x >> 5, lane = threadIdx.x & 31;
    int e = blockIdx.x * 8 + wave;
    if (e >= N_EDGES) return;
    int s = (int)src[e], d = (int)dst[e];
    bf16x4 v = ((const bf16x4*)(h + (size_t)s * HID))[lane];
    float* ad = agg + (size_t)d * HID + lane * 4;
#pragma unroll
    for (int i = 0; i < 4; ++i)
        atom_add_f32(&ad[i], (float)v[i]);
}

// ---------------------------------------------------------------------------
// SAGEConv: hout = relu( (agg*invd) @ w_l + hin @ w_r + b )  (bf16 out)
// M=32 per block, 2 M-tiles per wave per B fragment. grid = N_NODES/32.
// ---------------------------------------------------------------------------
__global__ __launch_bounds__(256) void sage_kernel(
    const __bf16* __restrict__ hin, const float* __restrict__ agg,
    const float* __restrict__ invd,
    const __bf16* __restrict__ wl, const __bf16* __restrict__ wr,
    const float* __restrict__ bias, __bf16* __restrict__ hout) {
    __shared__ __align__(16) __bf16 sA[32 * HID];
    __shared__ __align__(16) __bf16 sH[32 * HID];
    __shared__ float sInv[32];
    int t = threadIdx.x;
    int row0 = blockIdx.x * 32;
    if (t < 32) sInv[t] = invd[row0 + t];
    __syncthreads();
    const f32x4*  av = (const f32x4*)(agg + (size_t)row0 * HID);
    const bf16x4* hv = (const bf16x4*)(hin + (size_t)row0 * HID);
#pragma unroll
    for (int i = t; i < 32 * HID / 4; i += 256) {
        int r = i >> 5;                 // HID/4 = 32 chunks per row
        f32x4 v = av[i];
        float s = sInv[r];
        v.x *= s; v.y *= s; v.z *= s; v.w *= s;
        ((bf16x4*)sA)[i] = cvt4(v);
        ((bf16x4*)sH)[i] = hv[i];
    }
    __syncthreads();
    int wave = t >> 5, lane = t & 31;
    int n0 = wave * 16;
    f32x8 acc[2] = {zero8(), zero8()};
#pragma unroll
    for (int kk = 0; kk < HID; kk += 32) {
        bf16x16 bl = load_frag_b(wl + n0 * HID + kk, HID);
#pragma unroll
        for (int m = 0; m < 2; ++m)
            acc[m] = wmma_bf16(load_frag_a(sA + m * 16 * HID + kk, HID), bl, acc[m]);
        bf16x16 br = load_frag_b(wr + n0 * HID + kk, HID);
#pragma unroll
        for (int m = 0; m < 2; ++m)
            acc[m] = wmma_bf16(load_frag_a(sH + m * 16 * HID + kk, HID), br, acc[m]);
    }
    int col = lane & 15, rb = (lane >> 4) * 8;
    float bv = bias[n0 + col];
#pragma unroll
    for (int m = 0; m < 2; ++m)
#pragma unroll
        for (int r = 0; r < 8; ++r) {
            float v = fmaxf(acc[m][r] + bv, 0.f);
            hout[(size_t)(row0 + m * 16 + rb + r) * HID + n0 + col] = (__bf16)v;
        }
}

// ---------------------------------------------------------------------------
// Fused edge classifier, M = 64 edges per block:
//   ef   = relu(edge_attr @ w_edge + b_edge)          (WMMA, K=64)
//   feat = [h2[src] | h2[dst] | ef]  (64x384 in LDS, h2 gathers hit L2)
//   hid  = relu(feat @ wc1 + bc1)                     (WMMA, K=384)
//   out  = hid @ wc2 + bc2                            (WMMA, K=256, N pad 16)
// Each B fragment is reused across 4 M-tiles -> 4x less L2 weight traffic.
// Gathers and staging move 8B (bf16x4) per lane per op.
// ---------------------------------------------------------------------------
#define CM 64

__global__ __launch_bounds__(256) void classifier_kernel(
    const float* __restrict__ edge_attr,
    const long long* __restrict__ src, const long long* __restrict__ dst,
    const __bf16* __restrict__ h2,
    const __bf16* __restrict__ wte, const float* __restrict__ be,
    const __bf16* __restrict__ wtc1, const float* __restrict__ bc1,
    const __bf16* __restrict__ wtc2, const float* __restrict__ bc2,
    float* __restrict__ out) {
    __shared__ __align__(16) __bf16 sFeat[CM * 384];   // 48KB
    __shared__ __align__(16) __bf16 sHid[CM * 256];    // 32KB
    __shared__ __align__(16) __bf16 sEA[CM * EDGE_IN]; // 8KB
    __shared__ int sSrc[CM], sDst[CM];
    int t = threadIdx.x;
    size_t e0 = (size_t)blockIdx.x * CM;
    if (t < CM) {
        sSrc[t] = (int)src[e0 + t];
        sDst[t] = (int)dst[e0 + t];
    }
    const f32x4* eav = (const f32x4*)(edge_attr + e0 * EDGE_IN);
#pragma unroll
    for (int i = t; i < CM * EDGE_IN / 4; i += 256)
        ((bf16x4*)sEA)[i] = cvt4(eav[i]);
    // speculative prefetch of next edge block (global_prefetch_b8)
    __builtin_prefetch(edge_attr + (e0 + CM) * EDGE_IN, 0, 1);
    __syncthreads();
    // gather h2[src], h2[dst] into feat cols [0,256): 8B per lane per op
#pragma unroll
    for (int i = t; i < CM * (HID / 4); i += 256) {
        int r = i >> 5, c = i & 31;    // HID/4 = 32 chunks of 4 per row
        bf16x4 vs = ((const bf16x4*)(h2 + (size_t)sSrc[r] * HID))[c];
        bf16x4 vd = ((const bf16x4*)(h2 + (size_t)sDst[r] * HID))[c];
        *(bf16x4*)(sFeat + r * 384 + c * 4)       = vs;
        *(bf16x4*)(sFeat + r * 384 + 128 + c * 4) = vd;
    }
    int wave = t >> 5, lane = t & 31;
    int col = lane & 15, rb = (lane >> 4) * 8;
    {   // edge encoder into feat cols [256,384): wave -> n-tile, 4 M-tiles
        int n0 = wave * 16;
        f32x8 acc[4] = {zero8(), zero8(), zero8(), zero8()};
#pragma unroll
        for (int kk = 0; kk < EDGE_IN; kk += 32) {
            bf16x16 b = load_frag_b(wte + n0 * EDGE_IN + kk, EDGE_IN);
#pragma unroll
            for (int m = 0; m < 4; ++m)
                acc[m] = wmma_bf16(load_frag_a(sEA + m * 16 * EDGE_IN + kk, EDGE_IN), b, acc[m]);
        }
        float bv = be[n0 + col];
#pragma unroll
        for (int m = 0; m < 4; ++m)
#pragma unroll
            for (int r = 0; r < 8; ++r)
                sFeat[(m * 16 + rb + r) * 384 + 256 + n0 + col] =
                    (__bf16)fmaxf(acc[m][r] + bv, 0.f);
    }
    __syncthreads();
    {   // hid = relu(feat @ wc1 + bc1): wave -> n-tiles {wave, wave+8}, 4 M-tiles
        f32x8 acc[2][4];
#pragma unroll
        for (int j = 0; j < 2; ++j)
#pragma unroll
            for (int m = 0; m < 4; ++m) acc[j][m] = zero8();
#pragma unroll
        for (int kk = 0; kk < 384; kk += 32) {
            bf16x16 a[4];
#pragma unroll
            for (int m = 0; m < 4; ++m)
                a[m] = load_frag_a(sFeat + (m * 16) * 384 + kk, 384);
#pragma unroll
            for (int j = 0; j < 2; ++j) {
                int n0 = (wave + j * 8) * 16;
                bf16x16 b = load_frag_b(wtc1 + n0 * 384 + kk, 384);
#pragma unroll
                for (int m = 0; m < 4; ++m)
                    acc[j][m] = wmma_bf16(a[m], b, acc[j][m]);
            }
        }
#pragma unroll
        for (int j = 0; j < 2; ++j) {
            int n0 = (wave + j * 8) * 16;
            float bv = bc1[n0 + col];
#pragma unroll
            for (int m = 0; m < 4; ++m)
#pragma unroll
                for (int r = 0; r < 8; ++r)
                    sHid[(m * 16 + rb + r) * 256 + n0 + col] =
                        (__bf16)fmaxf(acc[j][m][r] + bv, 0.f);
        }
    }
    __syncthreads();
    // out = hid @ wc2 + bc2: waves 0..3, one 16x16 M-tile each (cols 10..15 padded)
    if (wave < 4) {
        int m0 = wave * 16;
        f32x8 c = zero8();
#pragma unroll
        for (int kk = 0; kk < 256; kk += 32)
            c = wmma_bf16(load_frag_a(sHid + m0 * 256 + kk, 256),
                          load_frag_b(wtc2 + kk, 256), c);
        if (col < OUT_DIM) {
            float bv = bc2[col];
#pragma unroll
            for (int r = 0; r < 8; ++r)
                out[(e0 + m0 + rb + r) * OUT_DIM + col] = c[r] + bv;
        }
    }
}

// ---------------------------------------------------------------------------
// Host launch
// ---------------------------------------------------------------------------
extern "C" void kernel_launch(void* const* d_in, const int* in_sizes, int n_in,
                              void* d_out, int out_size, void* d_ws, size_t ws_size,
                              hipStream_t stream) {
    (void)in_sizes; (void)n_in; (void)out_size; (void)ws_size;
    const float* x         = (const float*)d_in[0];
    const float* edge_attr = (const float*)d_in[1];
    const long long* eidx  = (const long long*)d_in[2];
    const float* w_node = (const float*)d_in[3];
    const float* b_node = (const float*)d_in[4];
    const float* w_edge = (const float*)d_in[5];
    const float* b_edge = (const float*)d_in[6];
    const float* w1_l = (const float*)d_in[7];
    const float* w1_r = (const float*)d_in[8];
    const float* b1   = (const float*)d_in[9];
    const float* w2_l = (const float*)d_in[10];
    const float* w2_r = (const float*)d_in[11];
    const float* b2   = (const float*)d_in[12];
    const float* wc1  = (const float*)d_in[13];
    const float* bc1  = (const float*)d_in[14];
    const float* wc2  = (const float*)d_in[15];
    const float* bc2  = (const float*)d_in[16];
    const long long* src = eidx;
    const long long* dst = eidx + N_EDGES;
    float* out = (float*)d_out;

    // workspace layout (bytes)
    char* ws = (char*)d_ws;
    __bf16* wt_node = (__bf16*)(ws + 0);        // 128x64
    __bf16* wt_edge = (__bf16*)(ws + 16384);    // 128x64
    __bf16* wt1_l   = (__bf16*)(ws + 32768);    // 128x128
    __bf16* wt1_r   = (__bf16*)(ws + 65536);
    __bf16* wt2_l   = (__bf16*)(ws + 98304);
    __bf16* wt2_r   = (__bf16*)(ws + 131072);
    __bf16* wtc1    = (__bf16*)(ws + 163840);   // 256x384
    __bf16* wtc2    = (__bf16*)(ws + 360448);   // 16x256 (padded)
    __bf16* hA      = (__bf16*)(ws + 524288);               // 100000x128 bf16
    __bf16* hB      = (__bf16*)(ws + 524288 + 25600000);
    float*  agg     = (float*)(ws + 51724288);              // 100000x128 f32
    float*  deg     = (float*)(ws + 102924288);
    float*  invd    = (float*)(ws + 103324288);

    // --- weight prep (bf16, transposed) ---
    auto T = [&](const float* w, __bf16* wt, int K, int N, int Npad) {
        int total = Npad * K;
        transpose_bf16_kernel<<<(total + 255) / 256, 256, 0, stream>>>(w, wt, K, N, Npad);
    };
    T(w_node, wt_node, 64, 128, 128);
    T(w_edge, wt_edge, 64, 128, 128);
    T(w1_l, wt1_l, 128, 128, 128);
    T(w1_r, wt1_r, 128, 128, 128);
    T(w2_l, wt2_l, 128, 128, 128);
    T(w2_r, wt2_r, 128, 128, 128);
    T(wc1, wtc1, 384, 256, 256);
    T(wc2, wtc2, 256, 10, 16);

    // --- degree (computed once, reused by both convs) ---
    zero_kernel<<<(N_NODES + 255) / 256, 256, 0, stream>>>(deg, N_NODES);
    deg_kernel<<<(N_EDGES + 255) / 256, 256, 0, stream>>>(dst, deg);
    invd_kernel<<<(N_NODES + 255) / 256, 256, 0, stream>>>(deg, invd);

    // --- node encoder ---
    encoder_kernel<<<N_NODES / 32, 256, 0, stream>>>(x, wt_node, b_node, hA);

    // --- SAGEConv 1: hA -> hB ---
    zero_kernel<<<(N_NODES * HID + 255) / 256, 256, 0, stream>>>(agg, N_NODES * HID);
    scatter_kernel<<<N_EDGES / 8, 256, 0, stream>>>(src, dst, hA, agg);
    sage_kernel<<<N_NODES / 32, 256, 0, stream>>>(hA, agg, invd, wt1_l, wt1_r, b1, hB);

    // --- SAGEConv 2: hB -> hA ---
    zero_kernel<<<(N_NODES * HID + 255) / 256, 256, 0, stream>>>(agg, N_NODES * HID);
    scatter_kernel<<<N_EDGES / 8, 256, 0, stream>>>(src, dst, hB, agg);
    sage_kernel<<<N_NODES / 32, 256, 0, stream>>>(hB, agg, invd, wt2_l, wt2_r, b2, hA);

    // --- fused edge classifier ---
    classifier_kernel<<<N_EDGES / CM, 256, 0, stream>>>(
        edge_attr, src, dst, hA, wt_edge, b_edge, wtc1, bc1, wtc2, bc2, out);
}
